// Memory_31928786878535
// MI455X (gfx1250) — matrix operand order repaired and checked by reading, not compiled, not run on hardware
//
#include <hip/hip_runtime.h>
#include <hip/hip_bf16.h>
#include <math.h>

typedef __attribute__((ext_vector_type(16))) _Float16 v16h;
typedef __attribute__((ext_vector_type(8)))  float    v8f;

#define DDIM   25088      // C*H*W = 512*7*7
#define NMEM   2000
#define BKEY   32
#define PI_2f  1.570795f
#define KSPLIT 8
#define KSEG   (DDIM / KSPLIT)   // 3136, multiple of 32

// Pack 16 fp32 (as 4x float4) into a v16h f16 fragment
__device__ __forceinline__ v16h pack16(float4 a, float4 b, float4 c, float4 d) {
    v16h r;
    r[0]=(_Float16)a.x;  r[1]=(_Float16)a.y;  r[2]=(_Float16)a.z;  r[3]=(_Float16)a.w;
    r[4]=(_Float16)b.x;  r[5]=(_Float16)b.y;  r[6]=(_Float16)b.z;  r[7]=(_Float16)b.w;
    r[8]=(_Float16)c.x;  r[9]=(_Float16)c.y;  r[10]=(_Float16)c.z; r[11]=(_Float16)c.w;
    r[12]=(_Float16)d.x; r[13]=(_Float16)d.y; r[14]=(_Float16)d.z; r[15]=(_Float16)d.w;
    return r;
}

// A-fragment (16x32 f16, M = lane&15): elements 0..7 -> K = kb+8*hi+i,
// elements 8..15 -> K = kb+16+8*hi+i  (ISA 7.12.2 16-bit A layout)
__device__ __forceinline__ v16h load_a_frag(const float* __restrict__ row,
                                            int kb, int hi) {
    const float4* a0 = (const float4*)(row + kb + 8 * hi);
    const float4* a1 = (const float4*)(row + kb + 16 + 8 * hi);
    return pack16(a0[0], a0[1], a1[0], a1[1]);
}

// ---------------------------------------------------------------------------
// Kernel 1: row norms of memory (2000 rows) and key (32 rows), clamped at eps
// ---------------------------------------------------------------------------
__global__ void norms_kernel(const float* __restrict__ key,
                             const float* __restrict__ mem,
                             float* __restrict__ m_norm,
                             float* __restrict__ k_norm) {
    int i   = blockIdx.x;
    int tid = threadIdx.x;
    const float* row = (i < NMEM) ? (mem + (size_t)i * DDIM)
                                  : (key + (size_t)(i - NMEM) * DDIM);
    float s = 0.f;
    for (int d = tid; d < DDIM; d += 256) {
        float v = row[d];
        s += v * v;
    }
    __shared__ float red[256];
    red[tid] = s;
    __syncthreads();
    for (int st = 128; st > 0; st >>= 1) {
        if (tid < st) red[tid] += red[tid + st];
        __syncthreads();
    }
    if (tid == 0) {
        float nv = fmaxf(sqrtf(red[0]), 1e-8f);
        if (i < NMEM) m_norm[i] = nv;
        else          k_norm[i - NMEM] = nv;
    }
}

// ---------------------------------------------------------------------------
// Kernel 2: zero the split-K dot accumulator (must be cleared every call)
// ---------------------------------------------------------------------------
__global__ void zero_kernel(float* __restrict__ p, int n) {
    int i = blockIdx.x * blockDim.x + threadIdx.x;
    if (i < n) p[i] = 0.f;
}

// ---------------------------------------------------------------------------
// Kernel 3: dots[b,n] = sum_d key[b,d]*mem[n,d]
// One wave per (nt, ksplit); computes the FULL M=32 (two 16x16 accumulators)
// so every memory element is loaded+converted exactly once per GEMM.
// ---------------------------------------------------------------------------
__global__ void dots_wmma_kernel(const float* __restrict__ key,
                                 const float* __restrict__ mem,
                                 float* __restrict__ dots) {
    int t  = blockIdx.x;
    int ks = t % KSPLIT;
    int nt = t / KSPLIT;                  // 0..124

    int lane = threadIdx.x;
    int col  = lane & 15;
    int hi   = lane >> 4;                 // 0 or 1

    const float* arow0 = key + (size_t)col * DDIM;          // key rows 0..15
    const float* arow1 = key + (size_t)(16 + col) * DDIM;   // key rows 16..31
    const float* brow  = mem + (size_t)(nt * 16 + col) * DDIM; // B: N = lane&15

    v8f acc0 = {};   // M = 0..15
    v8f acc1 = {};   // M = 16..31
    int k0 = ks * KSEG;
    for (int kt = 0; kt < KSEG; kt += 32) {
        int kb = k0 + kt;
        __builtin_prefetch(brow + kb + 256, 0, 0);  // global_prefetch_b8, ~1KB ahead

        // B fragment: elements 0..7 -> K = kb + 8*hi + i, elements 8..15 -> K = kb + 16 + 8*hi + i
        const float4* bp0 = (const float4*)(brow + kb + 8 * hi);
        const float4* bp1 = (const float4*)(brow + kb + 16 + 8 * hi);
        v16h b  = pack16(bp0[0], bp0[1], bp1[0], bp1[1]);
        v16h a0 = load_a_frag(arow0, kb, hi);
        v16h a1 = load_a_frag(arow1, kb, hi);

        acc0 = __builtin_amdgcn_wmma_f32_16x16x32_f16(false, a0, false, b,
                                                      (short)0, acc0, false, false);
        acc1 = __builtin_amdgcn_wmma_f32_16x16x32_f16(false, a1, false, b,
                                                      (short)0, acc1, false, false);
    }

    // C/D layout: lane L, vgpr r -> M = r + 8*hi, N = lane&15
    int n = nt * 16 + col;
    #pragma unroll
    for (int r = 0; r < 8; r++) {
        atomicAdd(&dots[(r + 8 * hi) * NMEM + n],        acc0[r]);
        atomicAdd(&dots[(16 + r + 8 * hi) * NMEM + n],   acc1[r]);
    }
}

// ---------------------------------------------------------------------------
// Kernel 4: w[b,:] = softmax_n( tan( (dots/(kn*mn)) * pi/2 ) )
// ---------------------------------------------------------------------------
__global__ void softmax_kernel(const float* __restrict__ dots,
                               const float* __restrict__ k_norm,
                               const float* __restrict__ m_norm,
                               float* __restrict__ w) {
    int b   = blockIdx.x;
    int tid = threadIdx.x;
    __shared__ float red[256];
    float kn = k_norm[b];
    float local[8];
    float mx = -1e30f;
    #pragma unroll
    for (int j = 0; j < 8; j++) {
        int n = tid + j * 256;
        float t = -1e30f;
        if (n < NMEM) {
            float cosv = dots[b * NMEM + n] / (kn * m_norm[n]);
            t = __tanf(cosv * PI_2f);
        }
        local[j] = t;
        mx = fmaxf(mx, t);
    }
    red[tid] = mx; __syncthreads();
    for (int s = 128; s > 0; s >>= 1) {
        if (tid < s) red[tid] = fmaxf(red[tid], red[tid + s]);
        __syncthreads();
    }
    mx = red[0]; __syncthreads();

    float sum = 0.f;
    #pragma unroll
    for (int j = 0; j < 8; j++) {
        int n = tid + j * 256;
        if (n < NMEM) {
            float e = __expf(local[j] - mx);
            local[j] = e;
            sum += e;
        }
    }
    red[tid] = sum; __syncthreads();
    for (int s = 128; s > 0; s >>= 1) {
        if (tid < s) red[tid] += red[tid + s];
        __syncthreads();
    }
    float inv = 1.f / red[0];
    #pragma unroll
    for (int j = 0; j < 8; j++) {
        int n = tid + j * 256;
        if (n < NMEM) w[b * NMEM + n] = local[j] * inv;
    }
}

// ---------------------------------------------------------------------------
// Kernel 5: out[b,d] = sum_n w[b,n] * mem[n,d]
// One wave per 16-wide column strip; full M=32 so the memory matrix streams
// once. The 32x16 fp32 B sub-tile is copied global->LDS with the CDNA5 async
// engine (GLOBAL_LOAD_ASYNC_TO_LDS_B128, ASYNCcnt), each lane owning one 64B
// row segment. Upper 16 rows are shifted by 32 floats (128B) so the two lane
// groups read disjoint bank windows (conflict-free), then converted f32->f16
// at LDS-read time to build the column-major B fragment.
// ---------------------------------------------------------------------------
__global__ void read_wmma_kernel(const float* __restrict__ mem,
                                 const float* __restrict__ w,
                                 float* __restrict__ out) {
    int nt = blockIdx.x;                  // 0..1567

    int lane = threadIdx.x;
    int col  = lane & 15;
    int hi   = lane >> 4;

    // 32 rows x 16 floats, rows 16..31 shifted +32 floats for bank decorrelation
    __shared__ float ldsf[32 * 16 + 32];

    const float* wrow0 = w + (size_t)col * NMEM;          // w rows 0..15
    const float* wrow1 = w + (size_t)(16 + col) * NMEM;   // w rows 16..31
    int dcol0 = nt * 16;

    // This lane's LDS row segment (64B, 16B-aligned incl. +128B shift)
    float* myrow = &ldsf[lane * 16 + (lane & 16) * 2];
    unsigned ldsoff = (unsigned)(uintptr_t)myrow;

    v8f acc0 = {};   // out rows 0..15
    v8f acc1 = {};   // out rows 16..31
    for (int kb = 0; kb < NMEM; kb += 32) {
        __syncthreads();   // all lanes done reading previous tile
        int grow = kb + lane;
        if (grow < NMEM) {
            unsigned long long ga =
                (unsigned long long)(uintptr_t)(mem + (size_t)grow * DDIM + dcol0);
            // Async DMA: LDS[ldsoff+off] = MEM[ga+off], 4x16B per lane
            asm volatile(
                "global_load_async_to_lds_b128 %0, %1, off\n\t"
                "global_load_async_to_lds_b128 %0, %1, off offset:16\n\t"
                "global_load_async_to_lds_b128 %0, %1, off offset:32\n\t"
                "global_load_async_to_lds_b128 %0, %1, off offset:48"
                :: "v"(ldsoff), "v"(ga) : "memory");
        } else {
            #pragma unroll
            for (int c = 0; c < 16; c++) myrow[c] = 0.f;   // K tail padding
        }
        asm volatile("s_wait_asynccnt 0" ::: "memory");
        __syncthreads();   // tile visible to all lanes

        // A fragments from w (fp32), zero-padded past K=2000
        v16h a0, a1;
        if (kb + 32 <= NMEM) {
            a0 = load_a_frag(wrow0, kb, hi);
            a1 = load_a_frag(wrow1, kb, hi);
        } else {
            #pragma unroll
            for (int i = 0; i < 8; i++) {
                int k0i = kb + 8 * hi + i;
                int k1i = kb + 16 + 8 * hi + i;
                a0[i]     = (k0i < NMEM) ? (_Float16)wrow0[k0i] : (_Float16)0.f;
                a0[8 + i] = (k1i < NMEM) ? (_Float16)wrow0[k1i] : (_Float16)0.f;
                a1[i]     = (k0i < NMEM) ? (_Float16)wrow1[k0i] : (_Float16)0.f;
                a1[8 + i] = (k1i < NMEM) ? (_Float16)wrow1[k1i] : (_Float16)0.f;
            }
        }

        // B fragment: element i -> K-row (i<8 ? 8*hi+i : 16+8*hi+i-8), N-col = lane&15
        v16h b;
        #pragma unroll
        for (int i = 0; i < 16; i++)
            b[i] = (_Float16)ldsf[((i < 8) ? (8 * hi + i) : (16 + 8 * hi + i - 8)) * 16 + ((i < 8) ? 0 : 32) + col];

        acc0 = __builtin_amdgcn_wmma_f32_16x16x32_f16(false, a0, false, b,
                                                      (short)0, acc0, false, false);
        acc1 = __builtin_amdgcn_wmma_f32_16x16x32_f16(false, a1, false, b,
                                                      (short)0, acc1, false, false);
    }

    #pragma unroll
    for (int r = 0; r < 8; r++) {
        out[(size_t)(r + 8 * hi) * DDIM + dcol0 + col]      = acc0[r];
        out[(size_t)(16 + r + 8 * hi) * DDIM + dcol0 + col] = acc1[r];
    }
}

// ---------------------------------------------------------------------------
extern "C" void kernel_launch(void* const* d_in, const int* in_sizes, int n_in,
                              void* d_out, int out_size, void* d_ws, size_t ws_size,
                              hipStream_t stream) {
    (void)in_sizes; (void)n_in; (void)out_size; (void)ws_size;
    const float* key = (const float*)d_in[0];   // [32, 512, 7, 7]
    const float* mem = (const float*)d_in[1];   // [2000, 512, 7, 7]
    float* out = (float*)d_out;                 // [32, 512, 7, 7]

    float* m_norm = (float*)d_ws;               // 2000
    float* k_norm = m_norm + NMEM;              // 32
    float* dots   = k_norm + BKEY;              // 32*2000
    float* w      = dots + BKEY * NMEM;         // 32*2000

    // 1) norms
    norms_kernel<<<NMEM + BKEY, 256, 0, stream>>>(key, mem, m_norm, k_norm);
    // 2) clear split-K accumulator
    zero_kernel<<<(BKEY * NMEM + 255) / 256, 256, 0, stream>>>(dots, BKEY * NMEM);
    // 3) dots GEMM (WMMA, full-M per wave, split-K=8): grid = 125 * 8
    dots_wmma_kernel<<<(NMEM / 16) * KSPLIT, 32, 0, stream>>>(key, mem, dots);
    // 4) tan-warp softmax
    softmax_kernel<<<BKEY, 256, 0, stream>>>(dots, k_norm, m_norm, w);
    // 5) weighted read GEMM (WMMA + async LDS staging): grid = 1568
    read_wmma_kernel<<<DDIM / 16, 32, 0, stream>>>(mem, w, out);
}
